// OneDConvertedParallelMHA2_24172075942158
// MI455X (gfx1250) — compile-verified, matrix operand order checked by reading, not run on hardware
//
#include <hip/hip_runtime.h>
#include <hip/hip_bf16.h>

// ---------------------------------------------------------------------------
// CDNA5 (gfx1250, wave32) bf16-WMMA implementation of causal GQA attention.
// Pipeline: convert/transpose weights -> fused QKV GEMM -> RoPE -> flash
// attention (WMMA QK^T and PV, online softmax) -> output GEMM.
// Global->LDS staging uses GLOBAL_LOAD_ASYNC_TO_LDS_B128 (ASYNCcnt) with a
// double-buffered LDS pipeline in the GEMMs.
// ---------------------------------------------------------------------------

typedef __bf16 bf16;
typedef __attribute__((ext_vector_type(16))) __bf16 v16bf;
typedef __attribute__((ext_vector_type(8)))  __bf16 v8bf;
typedef __attribute__((ext_vector_type(8)))  float  v8f;

#define WMMA_BF16(a, b, c) \
  __builtin_amdgcn_wmma_f32_16x16x32_bf16(false, (a), false, (b), (short)0, (c), false, false)

constexpr int Bb   = 2;
constexpr int S    = 2048;
constexpr int E    = 4096;
constexpr int H    = 32;
constexpr int HKV  = 8;
constexpr int Dh   = 128;
constexpr int MTOK = Bb * S;          // 4096 tokens
constexpr int NKV  = HKV * Dh;        // 1024
constexpr int NQKV = E + 2 * NKV;     // 6144 fused QKV columns

__device__ inline v8f zero8() {
  v8f z;
#pragma unroll
  for (int i = 0; i < 8; ++i) z[i] = 0.0f;
  return z;
}

__device__ inline v16bf pack16(v8bf lo, v8bf hi) {
  v16bf r;
#pragma unroll
  for (int i = 0; i < 8; ++i) { r[i] = lo[i]; r[i + 8] = hi[i]; }
  return r;
}

// Async DMA: global memory -> LDS, 16 bytes per lane, tracked by ASYNCcnt.
// LDS byte address = low 32 bits of the flat shared-memory pointer (ISA 10.2:
// LDS aperture uses addr[31:0] as the in-allocation offset).
__device__ inline void async_b128(void* lds_dst, const void* gsrc) {
  unsigned loff = (unsigned)(uintptr_t)lds_dst;
  unsigned long long g = (unsigned long long)(uintptr_t)gsrc;
  asm volatile("global_load_async_to_lds_b128 %0, %1, off"
               :: "v"(loff), "v"(g) : "memory");
}
__device__ inline void wait_async0() {
  asm volatile("s_wait_asynccnt 0" ::: "memory");
}

// ---------------------------------------------------------------------------
// Elementwise converts
// ---------------------------------------------------------------------------
__global__ void f32_to_bf16_kernel(const float* __restrict__ src,
                                   bf16* __restrict__ dst, int n) {
  int i = blockIdx.x * 256 + threadIdx.x;
  if (i < n) dst[i] = (bf16)src[i];
}

// src is K x N row-major f32; dst is N x K row-major bf16 (i.e. transposed)
__global__ void transpose_to_bf16_kernel(const float* __restrict__ src,
                                         bf16* __restrict__ dst, int K, int N) {
  int i = blockIdx.x * 256 + threadIdx.x;
  if (i < K * N) {
    int k = i / N, n = i - k * N;
    dst[(long)n * K + k] = (bf16)src[i];
  }
}

__global__ void bias_concat_kernel(const float* __restrict__ qb,
                                   const float* __restrict__ kb,
                                   const float* __restrict__ vb,
                                   float* __restrict__ dst) {
  int i = blockIdx.x * 256 + threadIdx.x;
  if (i < E)              dst[i] = qb[i];
  else if (i < E + NKV)   dst[i] = kb[i - E];
  else if (i < NQKV)      dst[i] = vb[i - E - NKV];
}

// ---------------------------------------------------------------------------
// Tiled bf16 WMMA GEMM with bias: C[M,N] = A[M,K] * BT[N,K]^T + bias[N]
// Block tile 128x128, 256 threads = 8 waves, each wave owns 32x64 (2x4 WMMA
// accumulators). K-step 32. Double-buffered LDS tiles filled by async DMA:
// tile k+1 streams in while WMMAs consume tile k.
// ---------------------------------------------------------------------------
template <bool OUT_BF16>
__global__ __launch_bounds__(256)
void gemm_bias_kernel(const bf16* __restrict__ A, const bf16* __restrict__ BT,
                      const float* __restrict__ bias, void* __restrict__ Cout,
                      int M, int N, int K) {
  constexpr int LDA = 40;  // 32 + 8 pad (keeps 16B alignment, dodges bank conflicts)
  __shared__ bf16 As[2][128 * LDA];
  __shared__ bf16 Bs[2][128 * LDA];

  const int t    = threadIdx.x;
  const int lane = t & 31;
  const int wave = t >> 5;
  const int wm   = wave >> 1;   // 0..3 : 32-row group
  const int wn   = wave & 1;    // 0..1 : 64-col group
  const int hi   = lane >> 4;   // half-wave select
  const int lr   = lane & 15;
  const int m0   = blockIdx.y * 128;
  const int n0   = blockIdx.x * 128;

  v8f acc[2][4];
#pragma unroll
  for (int i = 0; i < 2; ++i)
#pragma unroll
    for (int j = 0; j < 4; ++j) acc[i][j] = zero8();

  const int ldrow = t >> 1;        // 0..127
  const int ldk   = (t & 1) * 16;  // 0 or 16
  const bf16* agbase = A  + (long)(m0 + ldrow) * K + ldk;
  const bf16* bgbase = BT + (long)(n0 + ldrow) * K + ldk;

  // Prime the pipeline: tile k0=0 into buffer 0.
  async_b128(&As[0][ldrow * LDA + ldk],     agbase);
  async_b128(&As[0][ldrow * LDA + ldk + 8], agbase + 8);
  async_b128(&Bs[0][ldrow * LDA + ldk],     bgbase);
  async_b128(&Bs[0][ldrow * LDA + ldk + 8], bgbase + 8);

  int cur = 0;
  for (int k0 = 0; k0 < K; k0 += 32, cur ^= 1) {
    wait_async0();        // my async writes (this tile) landed in LDS
    __syncthreads();      // everyone's writes landed; prior readers done

    if (k0 + 32 < K) {    // stream next tile into the other buffer
      const bf16* ag = agbase + k0 + 32;
      const bf16* bg = bgbase + k0 + 32;
      __builtin_prefetch(ag + 32, 0, 1);
      __builtin_prefetch(bg + 32, 0, 1);
      async_b128(&As[cur ^ 1][ldrow * LDA + ldk],     ag);
      async_b128(&As[cur ^ 1][ldrow * LDA + ldk + 8], ag + 8);
      async_b128(&Bs[cur ^ 1][ldrow * LDA + ldk],     bg);
      async_b128(&Bs[cur ^ 1][ldrow * LDA + ldk + 8], bg + 8);
    }

    v16bf afr[2], bfr[4];
#pragma unroll
    for (int i = 0; i < 2; ++i) {
      int row = wm * 32 + i * 16 + lr;
      v8bf lo = *(const v8bf*)&As[cur][row * LDA + hi * 8];
      v8bf hh = *(const v8bf*)&As[cur][row * LDA + 16 + hi * 8];
      afr[i] = pack16(lo, hh);
    }
#pragma unroll
    for (int j = 0; j < 4; ++j) {
      int col = wn * 64 + j * 16 + lr;
      v8bf lo = *(const v8bf*)&Bs[cur][col * LDA + hi * 8];
      v8bf hh = *(const v8bf*)&Bs[cur][col * LDA + 16 + hi * 8];
      bfr[j] = pack16(lo, hh);
    }
#pragma unroll
    for (int i = 0; i < 2; ++i)
#pragma unroll
      for (int j = 0; j < 4; ++j)
        acc[i][j] = WMMA_BF16(afr[i], bfr[j], acc[i][j]);
  }

  // Epilogue: C layout — lane holds col (lr), VGPR r holds row r + 8*hi
#pragma unroll
  for (int i = 0; i < 2; ++i) {
#pragma unroll
    for (int j = 0; j < 4; ++j) {
      int n = n0 + wn * 64 + j * 16 + lr;
      float bv = bias[n];
#pragma unroll
      for (int r = 0; r < 8; ++r) {
        int m = m0 + wm * 32 + i * 16 + 8 * hi + r;
        float v = acc[i][j][r] + bv;
        if (OUT_BF16) ((bf16*)Cout)[(long)m * N + n] = (bf16)v;
        else          ((float*)Cout)[(long)m * N + n] = v;
      }
    }
  }
}

// ---------------------------------------------------------------------------
// RoPE with interleave-to-half: out[j] = x[2j]*cos - x[2j+1]*sin,
//                               out[j+64] = x[2j]*sin + x[2j+1]*cos
// One block = one (token, head); 64 threads, one j each. ROT == Dh == 128.
// ---------------------------------------------------------------------------
__global__ void rope_kernel(const bf16* __restrict__ qkv,
                            bf16* __restrict__ qo, bf16* __restrict__ ko) {
  const int tok  = blockIdx.x;       // 0..MTOK-1
  const int head = blockIdx.y;       // 0..H+HKV-1
  const int j    = threadIdx.x;      // 0..63
  const int s    = tok & (S - 1);    // sequence position (tok = b*S + s)

  // inv_freq = ROPE_BASE^(-2j/128); ln(10000) = 9.210340371976184
  float inv = __expf(-((float)(2 * j) * (1.0f / 128.0f)) * 9.210340371976184f);
  float ang = (float)s * inv;
  float c  = __cosf(ang);
  float sn = __sinf(ang);

  const bf16* src;
  bf16* dst;
  if (head < H) {
    src = qkv + (long)tok * NQKV + head * Dh;
    dst = qo  + (long)tok * E    + head * Dh;
  } else {
    int kh = head - H;
    src = qkv + (long)tok * NQKV + E + kh * Dh;
    dst = ko  + (long)tok * NKV  + kh * Dh;
  }
  float t1 = (float)src[2 * j];
  float t2 = (float)src[2 * j + 1];
  dst[j]      = (bf16)(t1 * c  - t2 * sn);
  dst[j + 64] = (bf16)(t1 * sn + t2 * c);
}

// ---------------------------------------------------------------------------
// Flash attention (causal, GQA rep=4). Block = 128 threads (4 waves), each
// wave owns 16 query rows; block covers 64 queries of one (b, h). KV chunked
// by 32 keys staged in LDS: K via async DMA (straight copy), V transposed
// manually for B-fragment loads. Q*K^T and P*V both run through
// v_wmma_f32_16x16x32_bf16; the probability tile is re-staged through LDS to
// convert C-layout -> A-fragment layout.
// ---------------------------------------------------------------------------
__global__ __launch_bounds__(128)
void attn_kernel(const bf16* __restrict__ q, const bf16* __restrict__ k,
                 const bf16* __restrict__ qkv, bf16* __restrict__ ctx) {
  constexpr int LKV = 136;  // Ks row stride: 128 d + 8 pad
  constexpr int LVT = 40;   // VsT row stride: 32 keys + 8 pad
  constexpr int LP  = 40;   // Ps row stride: 32 keys + 8 pad
  __shared__ bf16 Ks[32 * LKV];    // [key][d]
  __shared__ bf16 VsT[128 * LVT];  // [d][key]   (transposed on load)
  __shared__ bf16 Ps[4 * 16 * LP]; // per-wave P staging [wave][row][key]

  const int t    = threadIdx.x;
  const int lane = t & 31;
  const int wave = t >> 5;
  const int hi   = lane >> 4;
  const int lr   = lane & 15;
  const int b    = blockIdx.z;
  const int h    = blockIdx.y;
  const int q0   = blockIdx.x * 64;
  const int kvh  = h >> 2;  // H / HKV = 4

  const float scale = 0.08838834764831845f;  // 1/sqrt(128)
  const float LOG2E = 1.4426950408889634f;

  // Load this wave's Q fragments once (4 A-frags covering D=128).
  v16bf qf[4];
  {
    const int qrow = q0 + wave * 16 + lr;
    const bf16* qp = q + (long)(b * S + qrow) * E + h * Dh;
#pragma unroll
    for (int c = 0; c < 4; ++c) {
      v8bf lo = *(const v8bf*)(qp + c * 32 + hi * 8);
      v8bf hh = *(const v8bf*)(qp + c * 32 + 16 + hi * 8);
      qf[c] = pack16(lo, hh);
    }
  }

  v8f o[8];
#pragma unroll
  for (int j2 = 0; j2 < 8; ++j2) o[j2] = zero8();
  float mrow[8], lsum[8];
#pragma unroll
  for (int r = 0; r < 8; ++r) { mrow[r] = -3.0e38f; lsum[r] = 0.0f; }

  const int kend = q0 + 64;  // causal: keys <= q0+63 needed by this block
  for (int k0 = 0; k0 < kend; k0 += 32) {
    __syncthreads();  // previous chunk's Ks/VsT reads done before overwrite
    {
      // Cooperative load: 32 keys x 128 d. K chunk via async DMA, V chunk
      // transposed through VGPRs into LDS.
      int key  = t >> 2;          // 0..31
      int dseg = (t & 3) * 32;    // 0,32,64,96
      const bf16* kg = k   + (long)(b * S + k0 + key) * NKV  + kvh * Dh + dseg;
      const bf16* vg = qkv + (long)(b * S + k0 + key) * NQKV + E + NKV + kvh * Dh + dseg;
#pragma unroll
      for (int ss = 0; ss < 4; ++ss) {
        async_b128(&Ks[key * LKV + dseg + ss * 8], kg + ss * 8);
        v8bf vv = *(const v8bf*)(vg + ss * 8);
#pragma unroll
        for (int e = 0; e < 8; ++e)
          VsT[(dseg + ss * 8 + e) * LVT + key] = vv[e];
      }
    }
    wait_async0();
    __syncthreads();

    // scores = Q * K^T : two 16-key tiles, K reduced over D in 4 WMMA steps
    v8f sc[2];
#pragma unroll
    for (int j = 0; j < 2; ++j) {
      sc[j] = zero8();
#pragma unroll
      for (int c = 0; c < 4; ++c) {
        int krow = j * 16 + lr;
        v8bf lo = *(const v8bf*)&Ks[krow * LKV + c * 32 + hi * 8];
        v8bf hh = *(const v8bf*)&Ks[krow * LKV + c * 32 + 16 + hi * 8];
        sc[j] = WMMA_BF16(qf[c], pack16(lo, hh), sc[j]);
      }
    }

    // Online softmax: scale, causal mask, row max/sum via 16-lane shuffles.
#pragma unroll
    for (int r = 0; r < 8; ++r) {
      int qrow = q0 + wave * 16 + 8 * hi + r;
      float s0 = sc[0][r] * scale;
      float s1 = sc[1][r] * scale;
      if (k0 + lr      > qrow) s0 = -1.0e30f;
      if (k0 + 16 + lr > qrow) s1 = -1.0e30f;

      float mx = fmaxf(s0, s1);
      mx = fmaxf(mx, __shfl_xor(mx, 1, 32));
      mx = fmaxf(mx, __shfl_xor(mx, 2, 32));
      mx = fmaxf(mx, __shfl_xor(mx, 4, 32));
      mx = fmaxf(mx, __shfl_xor(mx, 8, 32));
      float mnew  = fmaxf(mrow[r], mx);
      float alpha = exp2f((mrow[r] - mnew) * LOG2E);
      mrow[r] = mnew;

      float p0 = exp2f((s0 - mnew) * LOG2E);
      float p1 = exp2f((s1 - mnew) * LOG2E);
      float rs = p0 + p1;
      rs += __shfl_xor(rs, 1, 32);
      rs += __shfl_xor(rs, 2, 32);
      rs += __shfl_xor(rs, 4, 32);
      rs += __shfl_xor(rs, 8, 32);
      lsum[r] = lsum[r] * alpha + rs;

      // Stage P (C layout -> LDS) for A-fragment reload.
      Ps[wave * 16 * LP + (8 * hi + r) * LP + lr]      = (bf16)p0;
      Ps[wave * 16 * LP + (8 * hi + r) * LP + 16 + lr] = (bf16)p1;

#pragma unroll
      for (int j2 = 0; j2 < 8; ++j2) o[j2][r] *= alpha;
    }

    // Same-wave LDS store->load ordering for Ps.
    asm volatile("s_wait_dscnt 0" ::: "memory");

    // Reload P as one A-fragment (16 rows x 32 keys).
    v16bf pa;
    {
      v8bf lo = *(const v8bf*)&Ps[wave * 16 * LP + lr * LP + hi * 8];
      v8bf hh = *(const v8bf*)&Ps[wave * 16 * LP + lr * LP + 16 + hi * 8];
      pa = pack16(lo, hh);
    }

    // O += P * V : 8 d-tiles of 16.
#pragma unroll
    for (int j2 = 0; j2 < 8; ++j2) {
      int drow = j2 * 16 + lr;
      v8bf lo = *(const v8bf*)&VsT[drow * LVT + hi * 8];
      v8bf hh = *(const v8bf*)&VsT[drow * LVT + 16 + hi * 8];
      o[j2] = WMMA_BF16(pa, pack16(lo, hh), o[j2]);
    }
  }

  // Normalize and emit ctx (bf16) in [B,S,E] layout.
#pragma unroll
  for (int j2 = 0; j2 < 8; ++j2) {
    int d = j2 * 16 + lr;
#pragma unroll
    for (int r = 0; r < 8; ++r) {
      int qrow = q0 + wave * 16 + 8 * hi + r;
      float v = o[j2][r] / lsum[r];
      ctx[(long)(b * S + qrow) * E + h * Dh + d] = (bf16)v;
    }
  }
}

// ---------------------------------------------------------------------------
// Host-side orchestration (single stream, graph-capture safe).
// ---------------------------------------------------------------------------
extern "C" void kernel_launch(void* const* d_in, const int* in_sizes, int n_in,
                              void* d_out, int out_size, void* d_ws, size_t ws_size,
                              hipStream_t stream) {
  const float* x    = (const float*)d_in[0];
  const float* wq_w = (const float*)d_in[1];
  const float* wq_b = (const float*)d_in[2];
  const float* wk_w = (const float*)d_in[3];
  const float* wk_b = (const float*)d_in[4];
  const float* wv_w = (const float*)d_in[5];
  const float* wv_b = (const float*)d_in[6];
  const float* wo_w = (const float*)d_in[7];
  const float* wo_b = (const float*)d_in[8];
  float* out = (float*)d_out;

  char* ws = (char*)d_ws;
  size_t off = 0;
  bf16* xb    = (bf16*)(ws + off); off += (size_t)MTOK * E    * 2;  // x in bf16
  bf16* btqkv = (bf16*)(ws + off); off += (size_t)NQKV * E    * 2;  // [wq;wk;wv]^T bf16
  bf16* wot   = (bf16*)(ws + off); off += (size_t)E    * E    * 2;  // wo^T bf16
  bf16* qkv   = (bf16*)(ws + off); off += (size_t)MTOK * NQKV * 2;  // fused QKV out
  bf16* qb    = (bf16*)(ws + off); off += (size_t)MTOK * E    * 2;  // Q after RoPE
  bf16* kb    = (bf16*)(ws + off); off += (size_t)MTOK * NKV  * 2;  // K after RoPE
  bf16* ctx   = (bf16*)(ws + off); off += (size_t)MTOK * E    * 2;  // attention out
  float* biascat = (float*)(ws + off); off += (size_t)NQKV * 4;

  // 1) Convert / transpose params to bf16.
  f32_to_bf16_kernel<<<(MTOK * E + 255) / 256, 256, 0, stream>>>(x, xb, MTOK * E);
  transpose_to_bf16_kernel<<<(E * E + 255) / 256, 256, 0, stream>>>(wq_w, btqkv, E, E);
  transpose_to_bf16_kernel<<<(E * NKV + 255) / 256, 256, 0, stream>>>(
      wk_w, btqkv + (size_t)E * E, E, NKV);
  transpose_to_bf16_kernel<<<(E * NKV + 255) / 256, 256, 0, stream>>>(
      wv_w, btqkv + (size_t)(E + NKV) * E, E, NKV);
  transpose_to_bf16_kernel<<<(E * E + 255) / 256, 256, 0, stream>>>(wo_w, wot, E, E);
  bias_concat_kernel<<<(NQKV + 255) / 256, 256, 0, stream>>>(wq_b, wk_b, wv_b, biascat);

  // 2) Fused QKV projection: [4096 x 4096] * [4096 x 6144] + bias.
  dim3 g1(NQKV / 128, MTOK / 128);
  gemm_bias_kernel<true><<<g1, 256, 0, stream>>>(xb, btqkv, biascat, qkv, MTOK, NQKV, E);

  // 3) RoPE on Q and K heads.
  dim3 g2(MTOK, H + HKV);
  rope_kernel<<<g2, 64, 0, stream>>>(qkv, qb, kb);

  // 4) Flash attention (V read directly from fused QKV buffer).
  dim3 g3(S / 64, H, Bb);
  attn_kernel<<<g3, 128, 0, stream>>>(qb, kb, qkv, ctx);

  // 5) Output projection to f32.
  dim3 g4(E / 128, MTOK / 128);
  gemm_bias_kernel<false><<<g4, 256, 0, stream>>>(ctx, wot, wo_b, out, MTOK, E, E);
}
